// BeeReceiver_50500225466645
// MI455X (gfx1250) — matrix operand order repaired and checked by reading, not compiled, not run on hardware
//
#include <hip/hip_runtime.h>
#include <hip/hip_bf16.h>

// ---------------------------------------------------------------------------
// Fused per-graph R-GCN for MI455X (gfx1250, wave32, WMMA).
// One workgroup (8 waves / 256 threads) per graph; all intermediates in LDS.
// Matmuls via v_wmma_f32_16x16x32_bf16. Edges counting-sorted by (rel, dst)
// in LDS; mean-aggregation is an atomic-free register gather per dst row.
// ---------------------------------------------------------------------------

typedef __attribute__((ext_vector_type(16))) __bf16 v16bf;
typedef __attribute__((ext_vector_type(8)))  __bf16 v8bf;
typedef __attribute__((ext_vector_type(4)))  __bf16 v4bf;
typedef __attribute__((ext_vector_type(2)))  __bf16 v2bf;
typedef __attribute__((ext_vector_type(8)))  float  v8f;

#define NPG      200      // nodes per graph
#define NPG_PAD  208      // padded to 13 * 16
#define EPG      3200     // edges per graph
#define FDIM     128
#define NREL     8
#define HS       136      // padded LDS row stride (bf16 elems) -> bank spread
#define MT       13       // M tiles
#define KT       4        // K steps of 32
#define BLOCK    256

// LDS partition (bytes)
#define OFF_HS    0
#define OFF_HW    (OFF_HS   + NPG_PAD * HS * 2)     // 56576
#define OFF_WS    (OFF_HW   + NPG_PAD * HS * 2)     // 113152
#define OFF_AGG   (OFF_WS   + FDIM    * HS * 2)     // 147968
#define OFF_CNT   (OFF_AGG  + NPG_PAD * FDIM * 4)   // 254464
#define OFF_MREP  (OFF_CNT  + NPG * NREL * 4)       // 260864
#define OFF_EORD  (OFF_MREP + FDIM * 4)             // 261376
#define OFF_TCNT  (OFF_EORD + EPG * 4)              // 274176
#define OFF_BOFF  (OFF_TCNT + NREL * 4)             // 274208
#define OFF_BCUR  (OFF_BOFF + NREL * (NPG + 1) * 4) // 280640
#define LDS_BYTES (OFF_BCUR + NREL * NPG * 4)       // 287040

// --- A fragment: 16x32 bf16 tile of h, row-major LDS (stride HS) -----------
// ISA 7.12.2: lane(0..15)=row M, lanes 16..31 same rows, K split hi/lo.
// VGPR p (p<4): K = hi*8 + 2p{,+1}; p>=4: +16.
__device__ inline v16bf load_A(const __bf16* h, int mt, int kk, int lane) {
  const int row  = mt * 16 + (lane & 15);
  const int hi   = lane >> 4;
  const __bf16* base = h + row * HS + kk * 32;
  v16bf a;
#pragma unroll
  for (int p = 0; p < 8; ++p) {
    const int kb = ((p >> 2) << 4) + (hi << 3) + ((p & 3) << 1);
    v2bf pr = *(const v2bf*)(base + kb);   // merged into ds_load_2addr_b32
    a[2 * p]     = pr[0];
    a[2 * p + 1] = pr[1];
  }
  return a;
}

// --- B fragment: 32x16 bf16 tile of W, staged *transposed* (w[n*HS + k]) ---
// lane(0..15)=col N; halfs j<8 -> K = kk*32 + hi*8 + j; j>=8 -> +16.
__device__ inline v16bf load_B(const __bf16* w, int nt, int kk, int lane) {
  const int col = nt * 16 + (lane & 15);
  const int hi  = lane >> 4;
  const v8bf* p = (const v8bf*)(w + col * HS + kk * 32 + hi * 8); // 2x ds_load_b128
  return __builtin_shufflevector(p[0], p[2],
                                 0, 1, 2, 3, 4, 5, 6, 7,
                                 8, 9, 10, 11, 12, 13, 14, 15);
}

extern "C" __global__ __launch_bounds__(BLOCK)
void rgcn_fused_kernel(const float* __restrict__ message,
                       const float* __restrict__ x,
                       const int*   __restrict__ ei,   // [2,E]
                       const int*   __restrict__ et,   // [E]
                       const float* __restrict__ W1, const float* __restrict__ root1,
                       const float* __restrict__ b1,
                       const float* __restrict__ W2, const float* __restrict__ root2,
                       const float* __restrict__ b2,
                       const float* __restrict__ embed,
                       const float* __restrict__ cont_w, const float* __restrict__ cont_b,
                       const float* __restrict__ msg_w,  const float* __restrict__ msg_b,
                       float* __restrict__ out, int Etot) {
  extern __shared__ char smem[];
  __bf16*   hs   = (__bf16*)  (smem + OFF_HS);    // [NPG_PAD][HS]  current h (bf16)
  __bf16*   hws  = (__bf16*)  (smem + OFF_HW);    // [NPG_PAD][HS]  h @ W_r
  __bf16*   ws   = (__bf16*)  (smem + OFF_WS);    // [FDIM][HS]     W_r transposed
  float*    agg  = (float*)   (smem + OFF_AGG);   // [NPG_PAD][FDIM]
  int*      cnt  = (int*)     (smem + OFF_CNT);   // [NPG][NREL]
  float*    mrep = (float*)   (smem + OFF_MREP);  // [FDIM]
  unsigned* eord = (unsigned*)(smem + OFF_EORD);  // [EPG] src | bf16norm<<16
  int*      tcnt = (int*)     (smem + OFF_TCNT);  // [NREL]
  int*      boff = (int*)     (smem + OFF_BOFF);  // [NREL][NPG+1] bucket offsets
  int*      bcur = (int*)     (smem + OFF_BCUR);  // [NREL][NPG]   bucket cursors

  const int g     = blockIdx.x;
  const int tid   = threadIdx.x;
  const int lane  = tid & 31;
  const int wave  = tid >> 5;
  const int ebase = g * EPG;
  const int nbase = g * NPG;

  // ---------------- Phase 0a: zero counters / pad rows ----------------------
  for (int i = tid; i < NPG * NREL; i += BLOCK) cnt[i] = 0;
  if (tid < NREL) tcnt[tid] = 0;
  for (int i = tid; i < 8 * HS; i += BLOCK) hs[NPG * HS + i] = (__bf16)0.0f;
  __syncthreads();

  // ---------------- Phase 0b: degree counts; x -> bf16 LDS; message repr ----
  for (int e = tid; e < EPG; e += BLOCK) {
    const int dl = ei[Etot + ebase + e] - nbase;
    const int ty = et[ebase + e];
    atomicAdd(&cnt[dl * NREL + ty], 1);
    atomicAdd(&tcnt[ty], 1);
  }

  for (int i = tid; i < NPG * (FDIM / 4); i += BLOCK) {
    const int row = i >> 5, c4 = i & 31;
    const float4 v = ((const float4*)x)[(size_t)(nbase + row) * 32 + c4];
    __bf16* d = hs + row * HS + c4 * 4;
    d[0] = (__bf16)v.x; d[1] = (__bf16)v.y; d[2] = (__bf16)v.z; d[3] = (__bf16)v.w;
  }

  // message repr: relu([emb(tok); relu(cont*w+b)] @ msg_w + msg_b)
  if (tid < FDIM) {
    const int   tok  = (int)message[g * 2 + 0];
    const float cont = message[g * 2 + 1];
    float s = msg_b[tid];
    for (int i = 0; i < FDIM; ++i)
      s += embed[tok * FDIM + i] * msg_w[i * FDIM + tid];
    for (int i = 0; i < FDIM; ++i) {
      float ce = cont * cont_w[i] + cont_b[i];
      ce = ce > 0.f ? ce : 0.f;
      s += ce * msg_w[(FDIM + i) * FDIM + tid];
    }
    mrep[tid] = s > 0.f ? s : 0.f;
  }
  __syncthreads();

  // ---------------- Phase 0c: per-(rel,dst) bucket offsets (wave32 scan) ----
  // Wave r scans the 200 per-dst counts of relation r; cursors written too.
  if (wave < NREL) {
    int base = 0;
    for (int j = 0; j < wave; ++j) base += tcnt[j];   // relation bucket start
    int* bo = boff + wave * (NPG + 1);
    int* bc = bcur + wave * NPG;
    for (int c = 0; c < NPG; c += 32) {
      const int dl = c + lane;
      const int v  = (dl < NPG) ? cnt[dl * NREL + wave] : 0;
      int x = v;
#pragma unroll
      for (int s = 1; s < 32; s <<= 1) {
        const int t = __shfl_up(x, s, 32);
        if (lane >= s) x += t;
      }
      if (dl < NPG) { bo[dl] = base + x - v; bc[dl] = base + x - v; }
      base += __shfl(x, 31, 32);
    }
    if (lane == 0) bo[NPG] = base;
  }
  __syncthreads();

  // ---------------- Phase 0d: counting sort by (rel, dst) -------------------
  // Edge word: src(8) | bf16(1/cnt)(16); dst is implicit from the bucket.
  for (int e = tid; e < EPG; e += BLOCK) {
    const int sl = ei[ebase + e] - nbase;
    const int dl = ei[Etot + ebase + e] - nbase;
    const int ty = et[ebase + e];
    const int cn = cnt[dl * NREL + ty];
    const float norm = 1.0f / (float)(cn > 0 ? cn : 1);
    const unsigned short nb = __builtin_bit_cast(unsigned short, (__bf16)norm);
    const int pos = atomicAdd(&bcur[ty * NPG + dl], 1);
    eord[pos] = (unsigned)sl | ((unsigned)nb << 16);
  }
  __syncthreads();

  // ---------------- Two R-GCN layers ---------------------------------------
  for (int L = 0; L < 2; ++L) {
    const float* Wl = L ? W2 : W1;
    const float* rl = L ? root2 : root1;
    const float* bl = L ? b2 : b1;

    for (int i = tid; i < NPG_PAD * FDIM; i += BLOCK) agg[i] = 0.f;
    __syncthreads();

    for (int r = 0; r < NREL; ++r) {
      // stage W_r transposed into LDS as bf16 (ws[n*HS + k])
      const float* wr = Wl + r * FDIM * FDIM;
      __builtin_prefetch(Wl + ((r + 1) & 7) * FDIM * FDIM, 0, 1);
      for (int i = tid; i < FDIM * FDIM / 4; i += BLOCK) {
        const float4 v = ((const float4*)wr)[i];
        const int k = (i * 4) >> 7, n = (i * 4) & 127;
        ws[(n + 0) * HS + k] = (__bf16)v.x;
        ws[(n + 1) * HS + k] = (__bf16)v.y;
        ws[(n + 2) * HS + k] = (__bf16)v.z;
        ws[(n + 3) * HS + k] = (__bf16)v.w;
      }
      __syncthreads();

      // GEMM: hws = hs @ W_r   (wave w owns column tile w)
      {
        const int nt = wave;
        v16bf bf[KT];
#pragma unroll
        for (int kk = 0; kk < KT; ++kk) bf[kk] = load_B(ws, nt, kk, lane);
        const int col = nt * 16 + (lane & 15);
        const int hi  = lane >> 4;
        for (int mt = 0; mt < MT; ++mt) {
          v8f c = {};
#pragma unroll
          for (int kk = 0; kk < KT; ++kk) {
            v16bf a = load_A(hs, mt, kk, lane);
            c = __builtin_amdgcn_wmma_f32_16x16x32_bf16(
                    false, a, false, bf[kk], (short)0, c, false, false);
          }
#pragma unroll
          for (int v = 0; v < 8; ++v)
            hws[(mt * 16 + v + 8 * hi) * HS + col] = (__bf16)c[v];
        }
      }
      __syncthreads();

      // gather this relation's messages per dst row (atomic-free):
      // wave w owns dst rows w, w+8, ...  ->  agg[dst] += sum(hws[src]*norm)
      {
        const int f0 = lane * 4;
        const int* bo = boff + r * (NPG + 1);
        for (int dl = wave; dl < NPG; dl += 8) {
          const int e0 = bo[dl], e1 = bo[dl + 1];
          if (e0 == e1) continue;
          float4 acc = make_float4(0.f, 0.f, 0.f, 0.f);
          for (int e = e0; e < e1; ++e) {
            const unsigned w = eord[e];               // broadcast ds_load_b32
            const int sl = (int)(w & 255u);
            const float norm =
                (float)__builtin_bit_cast(__bf16, (unsigned short)(w >> 16));
            const v4bf m = *(const v4bf*)(hws + sl * HS + f0);  // ds_load_b64
            acc.x += (float)m[0] * norm;
            acc.y += (float)m[1] * norm;
            acc.z += (float)m[2] * norm;
            acc.w += (float)m[3] * norm;
          }
          float4* arow = (float4*)(agg + dl * FDIM + f0);       // b128 RMW
          float4 a = *arow;
          a.x += acc.x; a.y += acc.y; a.z += acc.z; a.w += acc.w;
          *arow = a;
        }
      }
      __syncthreads();
    }

    // root term: agg += hs @ root  (WMMA with C loaded from agg)
    for (int i = tid; i < FDIM * FDIM / 4; i += BLOCK) {
      const float4 v = ((const float4*)rl)[i];
      const int k = (i * 4) >> 7, n = (i * 4) & 127;
      ws[(n + 0) * HS + k] = (__bf16)v.x;
      ws[(n + 1) * HS + k] = (__bf16)v.y;
      ws[(n + 2) * HS + k] = (__bf16)v.z;
      ws[(n + 3) * HS + k] = (__bf16)v.w;
    }
    __syncthreads();
    {
      const int nt  = wave;
      v16bf bf[KT];
#pragma unroll
      for (int kk = 0; kk < KT; ++kk) bf[kk] = load_B(ws, nt, kk, lane);
      const int col = nt * 16 + (lane & 15);
      const int hi  = lane >> 4;
      for (int mt = 0; mt < MT; ++mt) {
        v8f c;
#pragma unroll
        for (int v = 0; v < 8; ++v)
          c[v] = agg[(mt * 16 + v + 8 * hi) * FDIM + col];
#pragma unroll
        for (int kk = 0; kk < KT; ++kk) {
          v16bf a = load_A(hs, mt, kk, lane);
          c = __builtin_amdgcn_wmma_f32_16x16x32_bf16(
                  false, a, false, bf[kk], (short)0, c, false, false);
        }
#pragma unroll
        for (int v = 0; v < 8; ++v)
          agg[(mt * 16 + v + 8 * hi) * FDIM + col] = c[v];
      }
    }
    __syncthreads();

    // bias (+ReLU between layers); feed next layer's h
    if (L == 0) {
      for (int i = tid; i < NPG * FDIM; i += BLOCK) {
        const int n = i >> 7, f = i & 127;
        float v = agg[i] + bl[f];
        hs[n * HS + f] = (__bf16)(v > 0.f ? v : 0.f);
      }
    } else {
      for (int i = tid; i < NPG * FDIM; i += BLOCK) agg[i] += bl[i & 127];
    }
    __syncthreads();
  }

  // ---------------- Scores -> logits[g, p] ---------------------------------
  // batch is a uniform repeat of 200 nodes/graph, so pos == local node index
  // and every logits slot is written (no -inf padding survives).
  // Lane-parallel partials + wave32 shuffle reduction (bank-conflict free).
  for (int p = wave; p < NPG; p += 8) {
    const float* row = agg + p * FDIM;
    const int f0 = lane * 4;
    float s = 0.f;
#pragma unroll
    for (int k = 0; k < 4; ++k) s += row[f0 + k] * mrep[f0 + k];
#pragma unroll
    for (int m = 16; m; m >>= 1) s += __shfl_xor(s, m, 32);
    if (lane == 0) out[g * NPG + p] = s;
  }
}

extern "C" void kernel_launch(void* const* d_in, const int* in_sizes, int n_in,
                              void* d_out, int out_size, void* d_ws, size_t ws_size,
                              hipStream_t stream) {
  const float* message = (const float*)d_in[0];
  const float* x       = (const float*)d_in[1];
  const int*   ei      = (const int*)  d_in[2];
  const int*   et      = (const int*)  d_in[3];
  // d_in[4] = batch, d_in[5] = max_nodes: both implied by the uniform layout.
  const float* W1      = (const float*)d_in[6];
  const float* root1   = (const float*)d_in[7];
  const float* b1      = (const float*)d_in[8];
  const float* W2      = (const float*)d_in[9];
  const float* root2   = (const float*)d_in[10];
  const float* b2      = (const float*)d_in[11];
  const float* embed   = (const float*)d_in[12];
  const float* cont_w  = (const float*)d_in[13];
  const float* cont_b  = (const float*)d_in[14];
  const float* msg_w   = (const float*)d_in[15];
  const float* msg_b   = (const float*)d_in[16];

  const int Etot    = in_sizes[3];        // edges total
  const int nGraphs = in_sizes[0] / 2;    // message is [B, 2]

  // Idempotent, called every launch (no static state).
  hipFuncSetAttribute((const void*)rgcn_fused_kernel,
                      hipFuncAttributeMaxDynamicSharedMemorySize, LDS_BYTES);

  rgcn_fused_kernel<<<dim3(nGraphs), dim3(BLOCK), LDS_BYTES, stream>>>(
      message, x, ei, et, W1, root1, b1, W2, root2, b2,
      embed, cont_w, cont_b, msg_w, msg_b, (float*)d_out, Etot);
}
